// MultiheadAttention_12859132084800
// MI455X (gfx1250) — matrix operand order repaired and checked by reading, not compiled, not run on hardware
//
#include <hip/hip_runtime.h>
#include <hip/hip_bf16.h>

// ---------------------------------------------------------------------------
// Multihead self-attention for MI455X (gfx1250), bf16 WMMA pipeline.
//   T=2048, B=2, E=1024, H=16, hd=64
// ---------------------------------------------------------------------------

#define T_DIM   2048
#define B_DIM   2
#define E_DIM   1024
#define H_DIM   16
#define HD_DIM  64
#define M_DIM   (T_DIM * B_DIM)   // 4096 rows (t*B + b)
#define N_QKV   (3 * E_DIM)       // 3072
#define NEG_INF_VAL (-1e24f)

typedef __bf16 bf16_t;
typedef __attribute__((ext_vector_type(16))) __bf16 v16bf;
typedef __attribute__((ext_vector_type(8)))  __bf16 v8bf;
typedef __attribute__((ext_vector_type(8)))  float  v8f;

// D = A(16x32) * B(32x16) + C, bf16 inputs, fp32 accumulate
__device__ __forceinline__ v8f wmma_bf16(v16bf a, v16bf b, v8f c) {
  return __builtin_amdgcn_wmma_f32_16x16x32_bf16(false, a, false, b,
                                                 (short)0, c, false, false);
}

// A fragment (16x32, row-major, lda elements):
//   lane (m = lane&15, half = lane>>4), element e -> K = 16*half + e
// => two contiguous 16-byte loads per lane.
__device__ __forceinline__ v16bf load_frag_a(const bf16_t* tile, int lda) {
  const int lane = threadIdx.x & 31;
  const int m    = lane & 15;
  const int half = lane >> 4;
  const bf16_t* p = tile + m * lda + 16 * half;
  v8bf lo = *(const v8bf*)p;          // K = 16*half .. 16*half+7
  v8bf hi = *(const v8bf*)(p + 8);    // K = 16*half+8 .. 16*half+15
  return __builtin_shufflevector(lo, hi, 0, 1, 2, 3, 4, 5, 6, 7,
                                         8, 9, 10, 11, 12, 13, 14, 15);
}

// B fragment (32x16) sourced from an N-major matrix Bt[n][k] (K contiguous):
//   lane (n = lane&15, half = lane>>4), element e -> K = 16*half + e
// => one contiguous 32-byte load per lane.
__device__ __forceinline__ v16bf load_frag_b(const bf16_t* tile, int ldb) {
  const int lane = threadIdx.x & 31;
  const int n    = lane & 15;
  const int half = lane >> 4;
  return *(const v16bf*)(tile + n * ldb + 16 * half);
}

// ---------------------------------------------------------------------------
__global__ void cast_f32_to_bf16(const float* __restrict__ in,
                                 bf16_t* __restrict__ out, int n) {
  int i = blockIdx.x * blockDim.x + threadIdx.x;
  if (i < n) out[i] = (bf16_t)in[i];
}

// ---------------------------------------------------------------------------
// QKV projection: qkv[m,f] = sum_e x[m,e] * W[f,e] + bias[f]
// One wave (32 threads) computes a 64(M) x 32(N) tile: 8 accumulators.
// Epilogue scatters into attention layouts:
//   q_s [bh][t][d] (bf16, pre-scaled by hd^-0.5)
//   k_s [bh][t][d] (bf16)          (N-major for QK^T B-frags)
//   vt_s[bh][d][t] (bf16)          (N-major for P*V  B-frags)
// ---------------------------------------------------------------------------
__global__ void qkv_gemm_kernel(const bf16_t* __restrict__ xb,
                                const bf16_t* __restrict__ wib,
                                const float*  __restrict__ bias,
                                bf16_t* __restrict__ q_s,
                                bf16_t* __restrict__ k_s,
                                bf16_t* __restrict__ vt_s) {
  const int m0   = blockIdx.x * 64;
  const int n0   = blockIdx.y * 32;
  const int lane = threadIdx.x & 31;
  const int nl   = lane & 15;
  const int half = lane >> 4;

  v8f acc[4][2];
#pragma unroll
  for (int i = 0; i < 4; i++)
#pragma unroll
    for (int j = 0; j < 2; j++)
#pragma unroll
      for (int e = 0; e < 8; e++) acc[i][j][e] = 0.0f;

  for (int k0 = 0; k0 < E_DIM; k0 += 32) {
    v16bf a[4], b[2];
#pragma unroll
    for (int i = 0; i < 4; i++)
      a[i] = load_frag_a(xb + (size_t)(m0 + 16 * i) * E_DIM + k0, E_DIM);
#pragma unroll
    for (int j = 0; j < 2; j++)
      b[j] = load_frag_b(wib + (size_t)(n0 + 16 * j) * E_DIM + k0, E_DIM);
#pragma unroll
    for (int i = 0; i < 4; i++)
#pragma unroll
      for (int j = 0; j < 2; j++)
        acc[i][j] = wmma_bf16(a[i], b[j], acc[i][j]);
  }

#pragma unroll
  for (int j = 0; j < 2; j++) {
    const int f      = n0 + 16 * j + nl;
    const float bv   = bias[f];
    const int region = f >> 10;           // 0=q, 1=k, 2=v
    const int fr     = f & (E_DIM - 1);
    const int h      = fr >> 6;
    const int d      = fr & (HD_DIM - 1);
#pragma unroll
    for (int i = 0; i < 4; i++) {
#pragma unroll
      for (int r = 0; r < 8; r++) {
        const int m  = m0 + 16 * i + r + 8 * half;
        const int t  = m >> 1;            // B_DIM == 2
        const int bb = m & 1;
        const int hh = bb * H_DIM + h;
        const float val = acc[i][j][r] + bv;
        if (region == 0)
          q_s[((size_t)hh * T_DIM + t) * HD_DIM + d] = (bf16_t)(val * 0.125f);
        else if (region == 1)
          k_s[((size_t)hh * T_DIM + t) * HD_DIM + d] = (bf16_t)val;
        else
          vt_s[((size_t)hh * HD_DIM + d) * T_DIM + t] = (bf16_t)val;
      }
    }
  }
}

// ---------------------------------------------------------------------------
// Flash attention, one wave per (16-query tile, b*H+h).
// Scores: 16x32 per step (4 WMMA over hd=64); online softmax in fp32;
// P converted to bf16 through LDS to switch C-layout -> A-frag layout;
// P*V: 4 WMMA (N = hd = 64).  Output -> attn_s [(t*B+b)][e] bf16.
// ---------------------------------------------------------------------------
__global__ void attention_kernel(const bf16_t* __restrict__ q_s,
                                 const bf16_t* __restrict__ k_s,
                                 const bf16_t* __restrict__ vt_s,
                                 const unsigned char* __restrict__ mask,
                                 bf16_t* __restrict__ attn_s) {
  const int t0   = blockIdx.x * 16;
  const int hh   = blockIdx.y;          // b*H + h
  const int b    = hh >> 4;             // / H_DIM
  const int h    = hh & (H_DIM - 1);
  const int lane = threadIdx.x & 31;
  const int nl   = lane & 15;
  const int half = lane >> 4;

  __shared__ bf16_t Plds[16 * 32];      // P tile, row-major 16x32

  v16bf aq[2];
#pragma unroll
  for (int kk = 0; kk < 2; kk++)
    aq[kk] = load_frag_a(q_s + ((size_t)hh * T_DIM + t0) * HD_DIM + 32 * kk,
                         HD_DIM);

  v8f o[4];
  float mrun[8], lrun[8];
#pragma unroll
  for (int j = 0; j < 4; j++)
#pragma unroll
    for (int e = 0; e < 8; e++) o[j][e] = 0.0f;
#pragma unroll
  for (int r = 0; r < 8; r++) { mrun[r] = -__builtin_inff(); lrun[r] = 0.0f; }

  for (int s0 = 0; s0 < T_DIM; s0 += 32) {
    // ---- scores S = q * k^T   (16 queries x 32 keys) ----
    v8f sc[2];
#pragma unroll
    for (int jj = 0; jj < 2; jj++)
#pragma unroll
      for (int e = 0; e < 8; e++) sc[jj][e] = 0.0f;
#pragma unroll
    for (int jj = 0; jj < 2; jj++)
#pragma unroll
      for (int kk = 0; kk < 2; kk++)
        sc[jj] = wmma_bf16(
            aq[kk],
            load_frag_b(k_s + ((size_t)hh * T_DIM + s0 + 16 * jj) * HD_DIM +
                            32 * kk,
                        HD_DIM),
            sc[jj]);

    // ---- key padding mask (per column) ----
#pragma unroll
    for (int jj = 0; jj < 2; jj++) {
      const int s = s0 + 16 * jj + nl;
      if (mask[(size_t)s * B_DIM + b]) {
#pragma unroll
        for (int r = 0; r < 8; r++) sc[jj][r] = NEG_INF_VAL;
      }
    }

    // ---- online softmax (row stats reduced across the 16-lane half) ----
#pragma unroll
    for (int r = 0; r < 8; r++) {
      float bm = fmaxf(sc[0][r], sc[1][r]);
      bm = fmaxf(bm, __shfl_xor(bm, 1, 32));
      bm = fmaxf(bm, __shfl_xor(bm, 2, 32));
      bm = fmaxf(bm, __shfl_xor(bm, 4, 32));
      bm = fmaxf(bm, __shfl_xor(bm, 8, 32));
      const float mnew  = fmaxf(mrun[r], bm);
      const float alpha = __expf(mrun[r] - mnew);
      const float p0    = __expf(sc[0][r] - mnew);
      const float p1    = __expf(sc[1][r] - mnew);
      float rs = p0 + p1;
      rs += __shfl_xor(rs, 1, 32);
      rs += __shfl_xor(rs, 2, 32);
      rs += __shfl_xor(rs, 4, 32);
      rs += __shfl_xor(rs, 8, 32);
      lrun[r] = lrun[r] * alpha + rs;
      mrun[r] = mnew;
#pragma unroll
      for (int j = 0; j < 4; j++) o[j][r] *= alpha;
      // C-layout element (row r+8*half, col nl / nl+16) -> row-major LDS
      Plds[(r + 8 * half) * 32 + nl]      = (bf16_t)p0;
      Plds[(r + 8 * half) * 32 + 16 + nl] = (bf16_t)p1;
    }

    // LDS is wave-in-order, but fence the compiler and wait for stores.
    asm volatile("s_wait_dscnt 0x0" ::: "memory");

    // ---- O += P * V  (K = 32 keys, N = 64 = hd) ----
    v16bf ap = load_frag_a(Plds, 32);
#pragma unroll
    for (int j = 0; j < 4; j++)
      o[j] = wmma_bf16(
          ap,
          load_frag_b(vt_s + ((size_t)hh * HD_DIM + 16 * j) * T_DIM + s0,
                      T_DIM),
          o[j]);

    asm volatile("s_wait_dscnt 0x0" ::: "memory");  // WAR guard on Plds
  }

  // ---- finalize: divide by row sum, store bf16 [(t*B+b)][h*hd + d] ----
#pragma unroll
  for (int r = 0; r < 8; r++) {
    const int t     = t0 + r + 8 * half;
    const float inv = 1.0f / lrun[r];
#pragma unroll
    for (int j = 0; j < 4; j++) {
      attn_s[((size_t)t * B_DIM + b) * E_DIM + h * HD_DIM + 16 * j + nl] =
          (bf16_t)(o[j][r] * inv);
    }
  }
}

// ---------------------------------------------------------------------------
// Output projection: out[m,f] = sum_e attn[m,e] * out_w[f,e] + out_b[f]
// Same 64x32 wave tile as QKV GEMM; fp32 output straight to d_out.
// ---------------------------------------------------------------------------
__global__ void out_gemm_kernel(const bf16_t* __restrict__ attn_s,
                                const bf16_t* __restrict__ wob,
                                const float*  __restrict__ out_b,
                                float* __restrict__ out) {
  const int m0   = blockIdx.x * 64;
  const int n0   = blockIdx.y * 32;
  const int lane = threadIdx.x & 31;
  const int nl   = lane & 15;
  const int half = lane >> 4;

  v8f acc[4][2];
#pragma unroll
  for (int i = 0; i < 4; i++)
#pragma unroll
    for (int j = 0; j < 2; j++)
#pragma unroll
      for (int e = 0; e < 8; e++) acc[i][j][e] = 0.0f;

  for (int k0 = 0; k0 < E_DIM; k0 += 32) {
    v16bf a[4], b[2];
#pragma unroll
    for (int i = 0; i < 4; i++)
      a[i] = load_frag_a(attn_s + (size_t)(m0 + 16 * i) * E_DIM + k0, E_DIM);
#pragma unroll
    for (int j = 0; j < 2; j++)
      b[j] = load_frag_b(wob + (size_t)(n0 + 16 * j) * E_DIM + k0, E_DIM);
#pragma unroll
    for (int i = 0; i < 4; i++)
#pragma unroll
      for (int j = 0; j < 2; j++)
        acc[i][j] = wmma_bf16(a[i], b[j], acc[i][j]);
  }

#pragma unroll
  for (int j = 0; j < 2; j++) {
    const int f    = n0 + 16 * j + nl;
    const float bv = out_b[f];
#pragma unroll
    for (int i = 0; i < 4; i++)
#pragma unroll
      for (int r = 0; r < 8; r++) {
        const int m = m0 + 16 * i + r + 8 * half;
        out[(size_t)m * E_DIM + f] = acc[i][j][r] + bv;
      }
  }
}

// ---------------------------------------------------------------------------
extern "C" void kernel_launch(void* const* d_in, const int* in_sizes, int n_in,
                              void* d_out, int out_size, void* d_ws,
                              size_t ws_size, hipStream_t stream) {
  const float* x     = (const float*)d_in[0];  // [T,B,E]
  const float* w_in  = (const float*)d_in[1];  // [3E,E]
  const float* b_in  = (const float*)d_in[2];  // [3E]
  const float* w_out = (const float*)d_in[3];  // [E,E]
  const float* b_out = (const float*)d_in[4];  // [E]
  const unsigned char* mask = (const unsigned char*)d_in[5];  // [T,B] bool
  float* out = (float*)d_out;                  // [T,B,E]

  // ---- carve bf16 workspace (256B aligned sub-buffers) ----
  char* ws = (char*)d_ws;
  size_t off = 0;
  auto carve = [&](size_t elems) -> bf16_t* {
    bf16_t* p = (bf16_t*)(ws + off);
    off += (elems * sizeof(bf16_t) + 255) & ~(size_t)255;
    return p;
  };
  bf16_t* xb     = carve((size_t)M_DIM * E_DIM);                   // x bf16
  bf16_t* wib    = carve((size_t)N_QKV * E_DIM);                   // W_in bf16
  bf16_t* wob    = carve((size_t)E_DIM * E_DIM);                   // W_out bf16
  bf16_t* q_s    = carve((size_t)B_DIM * H_DIM * T_DIM * HD_DIM);  // q scaled
  bf16_t* k_s    = carve((size_t)B_DIM * H_DIM * T_DIM * HD_DIM);  // k
  bf16_t* vt_s   = carve((size_t)B_DIM * H_DIM * HD_DIM * T_DIM);  // v^T
  bf16_t* attn_s = carve((size_t)M_DIM * E_DIM);                   // attn out
  (void)ws_size; (void)in_sizes; (void)n_in; (void)out_size;

  // ---- 1) fp32 -> bf16 casts ----
  {
    int n = M_DIM * E_DIM;
    cast_f32_to_bf16<<<(n + 255) / 256, 256, 0, stream>>>(x, xb, n);
    n = N_QKV * E_DIM;
    cast_f32_to_bf16<<<(n + 255) / 256, 256, 0, stream>>>(w_in, wib, n);
    n = E_DIM * E_DIM;
    cast_f32_to_bf16<<<(n + 255) / 256, 256, 0, stream>>>(w_out, wob, n);
  }

  // ---- 2) QKV projection + scatter to attention layouts ----
  qkv_gemm_kernel<<<dim3(M_DIM / 64, N_QKV / 32), 32, 0, stream>>>(
      xb, wib, b_in, q_s, k_s, vt_s);

  // ---- 3) flash attention ----
  attention_kernel<<<dim3(T_DIM / 16, B_DIM * H_DIM), 32, 0, stream>>>(
      q_s, k_s, vt_s, mask, attn_s);

  // ---- 4) output projection ----
  out_gemm_kernel<<<dim3(M_DIM / 64, E_DIM / 32), 32, 0, stream>>>(
      attn_s, wob, b_out, out);
}
